// SAA_83786222010387
// MI455X (gfx1250) — matrix operand order repaired and checked by reading, not compiled, not run on hardware
//
#include <hip/hip_runtime.h>
#include <math.h>

// ---------------------------------------------------------------------------
// MHA forward for MI455X (gfx1250, wave32, WMMA).
//   B=4, S=2048, D=1024, H=16, DK=64.
// Pipeline (all on `stream`):
//   cvt_bf16   : X, Wq, Wk, Wv, Wo  -> bf16 once (GEMM loops become pure
//                load_b128 -> v_wmma, no per-iteration cvt VALU)
//   gemm<0>    : Q = X@Wq^T+bq -> bf16 [b,h,s,dk]
//   gemm<0>    : K             -> bf16 [b,h,s,dk]
//   gemm<1>    : V             -> bf16 [b,h,dk,s]  (transposed for P@V B-frags)
//   flash_attn : causal online-softmax attention -> AO bf16 [b,s,d]
//   gemm<2>    : out = AO@Wo^T+bo -> fp32 d_out
// Workspace: 5*16MB bf16 + 4*2MB bf16 = 88MB of d_ws.
// ---------------------------------------------------------------------------

typedef __attribute__((ext_vector_type(16))) __bf16 v16bf;
typedef __attribute__((ext_vector_type(8)))  __bf16 v8bf;
typedef __attribute__((ext_vector_type(8)))  float  v8f;

#define DEV __device__ __forceinline__

static constexpr int Bsz = 4, Ssz = 2048, Dsz = 1024, Hsz = 16, DKsz = 64;
static constexpr int Msz = Bsz * Ssz;      // 8192 tokens
static constexpr int Nsz = Dsz, Ksz = Dsz; // 1024

DEV v8f zero8() {
    v8f z;
#pragma unroll
    for (int i = 0; i < 8; ++i) z[i] = 0.0f;
    return z;
}

DEV v16bf cat8(v8bf a, v8bf b) {
    return __builtin_shufflevector(a, b, 0,1,2,3,4,5,6,7,8,9,10,11,12,13,14,15);
}

DEV v8f wmma_bf16(v16bf a, v16bf b, v8f c) {
    // D(f32 16x16) = A(bf16 16x32) * B(bf16 32x16) + C
    return __builtin_amdgcn_wmma_f32_16x16x32_bf16(false, a, false, b, (short)0, c,
                                                   false, false);
}

// ---------------------------------------------------------------------------
// fp32 -> bf16 bulk convert, 8 elements/thread (n must be a multiple of 2048)
// ---------------------------------------------------------------------------
__global__ __launch_bounds__(256) void cvt_bf16(const float* __restrict__ src,
                                                __bf16* __restrict__ dst) {
    const size_t i = ((size_t)blockIdx.x * 256 + threadIdx.x) * 8;
    const float4 f0 = *(const float4*)(src + i);
    const float4 f1 = *(const float4*)(src + i + 4);
    v8bf o;
    o[0]=(__bf16)f0.x; o[1]=(__bf16)f0.y; o[2]=(__bf16)f0.z; o[3]=(__bf16)f0.w;
    o[4]=(__bf16)f1.x; o[5]=(__bf16)f1.y; o[6]=(__bf16)f1.z; o[7]=(__bf16)f1.w;
    *(v8bf*)(dst + i) = o;
}

// ---------------------------------------------------------------------------
// GEMM: C[m,n] = sum_k A[m,k] * W[n,k] + bias[n]   (A @ W^T + b), A/W bf16.
// MODE 0: out bf16 at [b,h,s,dk]   (Q, K)
// MODE 1: out bf16 at [b,h,dk,s]   (V transposed)
// MODE 2: out fp32 at [m,n]        (output projection)
// Block: 256 thr = 8 waves as 2(M) x 4(N); wave tile 32x64 (8 wmma / k-step,
// B-frags shared across the two row tiles); block tile 64x256.
// Working set (X 16MB bf16 + W 2MB bf16) is L2-resident on the 192MB L2.
// ---------------------------------------------------------------------------
template <int MODE>
__global__ __launch_bounds__(256) void gemm_xwT(const __bf16* __restrict__ A,
                                                const __bf16* __restrict__ W,
                                                const float* __restrict__ bias,
                                                void* __restrict__ Outp) {
    const int tid  = threadIdx.x;
    const int lane = tid & 31, l15 = lane & 15, half = lane >> 4;
    const int wv = tid >> 5;
    const int wm = wv >> 2, wn = wv & 3;
    const int mBase = blockIdx.y * 64 + wm * 32;
    const int nBase = blockIdx.x * 256 + wn * 64;

    v8f acc0[4], acc1[4];
#pragma unroll
    for (int t = 0; t < 4; ++t) { acc0[t] = zero8(); acc1[t] = zero8(); }

    const __bf16* ar0 = A + (size_t)(mBase + l15) * Ksz;
    const __bf16* ar1 = A + (size_t)(mBase + 16 + l15) * Ksz;

    for (int kk = 0; kk < Ksz; kk += 32) {
        // A fragments: row m, K = {8*half..+7} U {16+8*half..+7}
        const v16bf af0 = cat8(*(const v8bf*)(ar0 + kk + half * 8),
                               *(const v8bf*)(ar0 + kk + 16 + half * 8));
        const v16bf af1 = cat8(*(const v8bf*)(ar1 + kk + half * 8),
                               *(const v8bf*)(ar1 + kk + 16 + half * 8));
#pragma unroll
        for (int t = 0; t < 4; ++t) {
            // B fragment: col n, K = {8*half..+7} U {16+8*half..+7} (row n of W)
            const int n = nBase + t * 16 + l15;
            const v16bf bf = cat8(*(const v8bf*)(W + (size_t)n * Ksz + kk + half * 8),
                                  *(const v8bf*)(W + (size_t)n * Ksz + kk + 16 + half * 8));
            acc0[t] = wmma_bf16(af0, bf, acc0[t]);
            acc1[t] = wmma_bf16(af1, bf, acc1[t]);
        }
    }

    // Epilogue: C layout -> row = tileBase + r + 8*half, col n = nBase+t*16+l15
#pragma unroll
    for (int mt = 0; mt < 2; ++mt) {
#pragma unroll
        for (int t = 0; t < 4; ++t) {
            const int n = nBase + t * 16 + l15;
            const float bvv = bias[n];
#pragma unroll
            for (int r = 0; r < 8; ++r) {
                const int m = mBase + mt * 16 + r + 8 * half;
                const float val = (mt ? acc1[t][r] : acc0[t][r]) + bvv;
                if constexpr (MODE == 2) {
                    ((float*)Outp)[(size_t)m * Nsz + n] = val;
                } else {
                    const int bb = m >> 11, ss = m & (Ssz - 1);   // m = bb*S+ss
                    const int hh = n >> 6,  dk = n & (DKsz - 1);  // n = hh*DK+dk
                    size_t idx;
                    if constexpr (MODE == 0)
                        idx = (((size_t)bb * Hsz + hh) * Ssz + ss) * DKsz + dk;
                    else
                        idx = (((size_t)bb * Hsz + hh) * DKsz + dk) * Ssz + ss;
                    ((__bf16*)Outp)[idx] = (__bf16)val;
                }
            }
        }
    }
}

// ---------------------------------------------------------------------------
// Flash attention, one wave per 16-query tile of one (b,h).
// Computes S^T tiles (A=K rows, B=Q cols): lane = query, VGPR r = key r+8*half.
// That C-layout IS the A-fragment layout for the P@V WMMA -> no transpose.
// Online softmax state is per-lane (query = lane&15), merged across the
// lane<->lane+16 pair with shfl_xor; O rescale factors broadcast with shfl.
// ---------------------------------------------------------------------------
__global__ __launch_bounds__(256) void flash_attn(const __bf16* __restrict__ Q,
                                                  const __bf16* __restrict__ Km,
                                                  const __bf16* __restrict__ VT,
                                                  const float* __restrict__ rw,
                                                  const int* __restrict__ maskp,
                                                  __bf16* __restrict__ Oo) {
    const int tid  = threadIdx.x;
    const int lane = tid & 31, l15 = lane & 15, half = lane >> 4;
    const int wave = blockIdx.x * 8 + (tid >> 5);

    constexpr int QT = Ssz / 16;              // 128 q-tiles per (b,h)
    const int b     = wave / (Hsz * QT);
    const int rem   = wave - b * Hsz * QT;
    const int h     = rem / QT;
    const int qBase = (rem - h * QT) * 16;

    const __bf16* Qb  = Q  + ((size_t)(b * Hsz + h) * Ssz) * DKsz;
    const __bf16* Kb  = Km + ((size_t)(b * Hsz + h) * Ssz) * DKsz;
    const __bf16* Vb  = VT + ((size_t)(b * Hsz + h) * DKsz) * Ssz;
    const float*  rwp = rw + (size_t)b * Ssz;

    const int causal = maskp[0];
    const int q_lane = qBase + l15;

    // Q B-fragments (d-blocks 0,1): col = q_lane, d = {8*half..+7} U {16+8*half..+7}
    const v16bf bq0 = cat8(*(const v8bf*)(Qb + (size_t)q_lane * DKsz + half * 8),
                           *(const v8bf*)(Qb + (size_t)q_lane * DKsz + 16 + half * 8));
    const v16bf bq1 = cat8(*(const v8bf*)(Qb + (size_t)q_lane * DKsz + 32 + half * 8),
                           *(const v8bf*)(Qb + (size_t)q_lane * DKsz + 48 + half * 8));

    float m_run = -INFINITY, l_run = 0.0f;
    v8f oacc[4];
#pragma unroll
    for (int t = 0; t < 4; ++t) oacc[t] = zero8();

    int kEnd = causal ? (qBase + 16) : Ssz;
    if (kEnd > Ssz) kEnd = Ssz;
    const int numK = (kEnd + 31) & ~31;       // <= Ssz always

    for (int kb = 0; kb < numK; kb += 32) {
        // --- S^T = K * Q^T : two 16(key)x16(query) tiles ---
        v8f st0 = zero8(), st1 = zero8();
        {
            const __bf16* kr0 = Kb + (size_t)(kb + l15) * DKsz;
            v16bf ak = cat8(*(const v8bf*)(kr0 + half * 8),
                            *(const v8bf*)(kr0 + 16 + half * 8));
            st0 = wmma_bf16(ak, bq0, st0);
            ak = cat8(*(const v8bf*)(kr0 + 32 + half * 8),
                      *(const v8bf*)(kr0 + 48 + half * 8));
            st0 = wmma_bf16(ak, bq1, st0);
        }
        {
            const __bf16* kr1 = Kb + (size_t)(kb + 16 + l15) * DKsz;
            v16bf ak = cat8(*(const v8bf*)(kr1 + half * 8),
                            *(const v8bf*)(kr1 + 16 + half * 8));
            st1 = wmma_bf16(ak, bq0, st1);
            ak = cat8(*(const v8bf*)(kr1 + 32 + half * 8),
                      *(const v8bf*)(kr1 + 48 + half * 8));
            st1 = wmma_bf16(ak, bq1, st1);
        }

        // per-key reweight values (8 consecutive keys per tile per lane)
        const float4 ra0 = *(const float4*)(rwp + kb + 8 * half);
        const float4 ra1 = *(const float4*)(rwp + kb + 8 * half + 4);
        const float4 rb0 = *(const float4*)(rwp + kb + 16 + 8 * half);
        const float4 rb1 = *(const float4*)(rwp + kb + 16 + 8 * half + 4);
        const float rw0[8] = {ra0.x, ra0.y, ra0.z, ra0.w, ra1.x, ra1.y, ra1.z, ra1.w};
        const float rw1[8] = {rb0.x, rb0.y, rb0.z, rb0.w, rb1.x, rb1.y, rb1.z, rb1.w};

        // scores: (qk - mask*1e9) * (1/sqrt(DK)) * rw[key]   (reference order)
        float p0[8], p1[8];
        float bm = -INFINITY;
#pragma unroll
        for (int r = 0; r < 8; ++r) {
            const int key0 = kb + r + 8 * half;
            const int key1 = key0 + 16;
            float s0 = st0[r], s1 = st1[r];
            if (causal) {
                if (key0 > q_lane) s0 -= 1e9f;
                if (key1 > q_lane) s1 -= 1e9f;
            }
            s0 = s0 * 0.125f * rw0[r];
            s1 = s1 * 0.125f * rw1[r];
            p0[r] = s0; p1[r] = s1;
            bm = fmaxf(bm, fmaxf(s0, s1));
        }
        bm = fmaxf(bm, __shfl_xor(bm, 16, 32));      // merge key halves per query
        const float m_new = fmaxf(m_run, bm);
        const float fl    = __expf(m_run - m_new);

        float ps = 0.0f;
#pragma unroll
        for (int r = 0; r < 8; ++r) {
            p0[r] = __expf(p0[r] - m_new);
            p1[r] = __expf(p1[r] - m_new);
            ps += p0[r] + p1[r];
        }
        l_run = l_run * fl + ps;
        m_run = m_new;

        // rescale O: row r+8*half is query (r+8*half); its factor lives there
#pragma unroll
        for (int r = 0; r < 8; ++r) {
            const float fr = __shfl(fl, r + 8 * half, 32);
#pragma unroll
            for (int t = 0; t < 4; ++t) oacc[t][r] *= fr;
        }

        // P fragment: C-layout of S^T == A-layout of P (lane=query row,
        // keys {8*half..+7} from tile0, {16+8*half..+7} from tile1)
        v16bf ap;
#pragma unroll
        for (int r = 0; r < 8; ++r) {
            ap[r]     = (__bf16)p0[r];
            ap[8 + r] = (__bf16)p1[r];
        }

        // O += P @ V : B-frag from V^T, col d, keys {8*half..+7} U {16+8*half..+7}
#pragma unroll
        for (int dt = 0; dt < 4; ++dt) {
            const int d = dt * 16 + l15;
            const v16bf bv = cat8(*(const v8bf*)(Vb + (size_t)d * Ssz + kb + half * 8),
                                  *(const v8bf*)(Vb + (size_t)d * Ssz + kb + 16 + half * 8));
            oacc[dt] = wmma_bf16(ap, bv, oacc[dt]);
        }
    }

    // finalize: total row sum across the lane pair, normalize, store [B,S,D]
    const float l_tot = l_run + __shfl_xor(l_run, 16, 32);
    const float linv  = 1.0f / l_tot;
#pragma unroll
    for (int r = 0; r < 8; ++r) {
        const float lr  = __shfl(linv, r + 8 * half, 32);
        const int  qrow = qBase + r + 8 * half;
        const size_t base = ((size_t)b * Ssz + qrow) * Dsz + h * DKsz;
#pragma unroll
        for (int dt = 0; dt < 4; ++dt)
            Oo[base + dt * 16 + l15] = (__bf16)(oacc[dt][r] * lr);
    }
}

// ---------------------------------------------------------------------------
extern "C" void kernel_launch(void* const* d_in, const int* in_sizes, int n_in,
                              void* d_out, int out_size, void* d_ws, size_t ws_size,
                              hipStream_t stream) {
    (void)in_sizes; (void)n_in; (void)out_size; (void)ws_size;

    const float* X     = (const float*)d_in[0];
    const float* rw    = (const float*)d_in[1];
    const int*   maskp = (const int*)d_in[2];
    const float* Wq    = (const float*)d_in[3];
    const float* bq    = (const float*)d_in[4];
    const float* Wk    = (const float*)d_in[5];
    const float* bk    = (const float*)d_in[6];
    const float* Wv    = (const float*)d_in[7];
    const float* bv    = (const float*)d_in[8];
    const float* Wo    = (const float*)d_in[9];
    const float* bo    = (const float*)d_in[10];
    float* out = (float*)d_out;

    const size_t E  = (size_t)Bsz * Ssz * Dsz;   // 8388608 activation elements
    const size_t WE = (size_t)Dsz * Dsz;         // 1048576 weight elements
    __bf16* q   = (__bf16*)d_ws;                 // 16MB each
    __bf16* k   = q + E;
    __bf16* vt  = k + E;
    __bf16* ao  = vt + E;
    __bf16* xb  = ao + E;
    __bf16* wqb = xb + E;                        // 2MB each
    __bf16* wkb = wqb + WE;
    __bf16* wvb = wkb + WE;
    __bf16* wob = wvb + WE;

    // one-time-per-call fp32 -> bf16 packs (bandwidth-trivial)
    cvt_bf16<<<(int)(E / 2048),  256, 0, stream>>>(X,  xb);
    cvt_bf16<<<(int)(WE / 2048), 256, 0, stream>>>(Wq, wqb);
    cvt_bf16<<<(int)(WE / 2048), 256, 0, stream>>>(Wk, wkb);
    cvt_bf16<<<(int)(WE / 2048), 256, 0, stream>>>(Wv, wvb);
    cvt_bf16<<<(int)(WE / 2048), 256, 0, stream>>>(Wo, wob);

    dim3 gP(Nsz / 256, Msz / 64);                // (4, 128)
    gemm_xwT<0><<<gP, 256, 0, stream>>>(xb, wqb, bq, (void*)q);
    gemm_xwT<0><<<gP, 256, 0, stream>>>(xb, wkb, bk, (void*)k);
    gemm_xwT<1><<<gP, 256, 0, stream>>>(xb, wvb, bv, (void*)vt);

    const int nWaves = Bsz * Hsz * (Ssz / 16);   // 8192
    flash_attn<<<nWaves / 8, 256, 0, stream>>>(q, k, vt, rw, maskp, ao);

    gemm_xwT<2><<<gP, 256, 0, stream>>>(ao, wob, bo, (void*)out);
}
